// CEncoderFiLM_38963943309884
// MI455X (gfx1250) — matrix operand order repaired and checked by reading, not compiled, not run on hardware
//
#include <hip/hip_runtime.h>
#include <hip/hip_bf16.h>
#include <cstdint>
#include <cstddef>

// ---------------------------------------------------------------------------
// Types for gfx1250 WMMA
// ---------------------------------------------------------------------------
typedef __attribute__((ext_vector_type(8)))  _Float16 v8h;
typedef __attribute__((ext_vector_type(16))) _Float16 v16h;
typedef __attribute__((ext_vector_type(8)))  float    v8f;

union V16U { v16h v; v8h h[2]; };
static __device__ __forceinline__ v16h make16(v8h lo, v8h hi) {
    V16U u; u.h[0] = lo; u.h[1] = hi; return u.v;
}

static __device__ __forceinline__ float2 cmulacc(float2 a, float2 b, float2 c) {
    c.x = fmaf(a.x, b.x, fmaf(-a.y, b.y, c.x));
    c.y = fmaf(a.x, b.y, fmaf( a.y, b.x, c.y));
    return c;
}

// ---------------------------------------------------------------------------
// Weight transform: complex (cout,cin,3,3) fp32 -> f16 A matrix [2*cout][Kpad]
// k order: k = tap*(2*cin) + r*cin + ic
// row m<cout  (real out): r=0 -> wr, r=1 -> -wi
// row m>=cout (imag out): r=0 -> wi, r=1 ->  wr
// ---------------------------------------------------------------------------
__global__ void wtrans_kernel(const float* __restrict__ w, _Float16* __restrict__ Aw,
                              int lc, int cout, int K, int Kpad) {
    int k = blockIdx.x * 64 + threadIdx.x;
    int m = blockIdx.y;
    int cin = 1 << lc;
    float val = 0.f;
    if (k < K) {
        int tap = k >> (lc + 1);
        int rc  = (k >> lc) & 1;
        int ic  = k & (cin - 1);
        int im  = (m >= cout) ? 1 : 0;
        int ch  = m - im * cout;
        size_t widx = ((size_t)(ch * cin + ic) * 9 + tap) * 2;
        float wr = w[widx], wi = w[widx + 1];
        val = im ? (rc ? wr : wi) : (rc ? -wi : wr);
    }
    Aw[(size_t)m * Kpad + k] = (_Float16)val;
}

// ---------------------------------------------------------------------------
// k-decode table: ktab[k] = { float offset within sample, tap index (9=pad) }
// float offset = (ic << (lihw+1)) + (kh << (liw+1)) + (kw << 1) + rc
// ---------------------------------------------------------------------------
__global__ void ktab_kernel(int2* __restrict__ ktab, int lc, int liw, int lihw,
                            int K, int Kpad) {
    int k = blockIdx.x * 64 + threadIdx.x;
    if (k >= Kpad) return;
    int tap = 9, off = 0;
    if (k < K) {
        tap = k >> (lc + 1);
        int rc = (k >> lc) & 1;
        int ic = k & ((1 << lc) - 1);
        int kh = (tap * 11) >> 5;          // tap/3 for tap<10
        int kw = tap - kh * 3;
        off = (ic << (lihw + 1)) + (kh << (liw + 1)) + (kw << 1) + rc;
    }
    ktab[k] = make_int2(off, tap);
}

// ---------------------------------------------------------------------------
// Implicit-GEMM complex conv with WMMA f32_16x16x32_f16.
// Tile: M=64 x N=128 per 256-thread WG (8 waves: 4 mi x 2 ni). Each wave:
// one 16-row A fragment reused by FOUR 16-col B fragments -> 4 WMMAs/chunk.
// B patch tile DOUBLE-BUFFERED in LDS [n][k] (40-half rows, 16B aligned):
// one barrier per chunk, next chunk's gathers overlap current chunk's WMMAs.
// ktab is brought into LDS via CDNA5 async global->LDS loads (ASYNCcnt).
// ---------------------------------------------------------------------------
__global__ __launch_bounds__(256) void conv_wmma_kernel(
    const float* __restrict__ xin,      // complex interleaved input
    const _Float16* __restrict__ Aw,    // f16 weights [2*cout][Kpad]
    const int2* __restrict__ ktabG,     // k-decode table [Kpad]
    float* __restrict__ yout,           // complex interleaved output segment
    int lc, int stride, int liw, int lihw, int low, int lohw,
    int cout, int Kpad)
{
    const int IW = 1 << liw;
    const int IH = IW;
    const int OW = 1 << low;

    __shared__ __align__(16) _Float16 ldsB0[128 * 40];  // 10 KB
    __shared__ __align__(16) _Float16 ldsB1[128 * 40];  // 10 KB
    __shared__ int2 ldsK[4608];                         // 36 KB (max Kpad)

    const int tid  = threadIdx.x;
    const int lane = tid & 31;
    const int wave = tid >> 5;
    const int mi = wave >> 1;          // 0..3
    const int ni = wave & 1;           // 0..1
    const int mTile  = blockIdx.y * 64 + mi * 16;
    const int nTile0 = blockIdx.x * 128;

    // ---- async copy ktab -> LDS (gfx1250 ASYNCcnt path, no VGPR data) -----
    for (int j = tid; j < Kpad; j += 256) {
        unsigned lofs = (unsigned)(size_t)(&ldsK[j]);   // wg-relative LDS byte addr
        const int2* g = ktabG + j;
        asm volatile("global_load_async_to_lds_b64 %0, %1, off"
                     :: "v"(lofs), "v"(g) : "memory");
    }
    asm volatile("s_wait_asynccnt 0x0" ::: "memory");
    __syncthreads();

    // --- staging thread: fixed spatial decode (each thread owns one n row) --
    const int nn_f = tid >> 1;               // 0..127
    const int kk_f = (tid & 1) << 4;         // 0 or 16
    const int n_f  = nTile0 + nn_f;
    const int ow_f = n_f & (OW - 1);
    const int oh_f = (n_f >> low) & (OW - 1);
    const int b_f  = n_f >> lohw;
    const int ih0  = oh_f * stride - 1;
    const int iw0  = ow_f * stride - 1;
    const int baseF = (b_f << (lc + lihw + 1)) + ih0 * (IW * 2) + iw0 * 2;
    // 3x3 halo validity bitmask (bit kh*3+kw); padding taps (9+) read 0
    int vmask = 0;
    #pragma unroll
    for (int kh = 0; kh < 3; ++kh)
        #pragma unroll
        for (int kw = 0; kw < 3; ++kw) {
            int ih = ih0 + kh, iw = iw0 + kw;
            if ((unsigned)ih < (unsigned)IH && (unsigned)iw < (unsigned)IW)
                vmask |= 1 << (kh * 3 + kw);
        }

    auto stage = [&](int k0, _Float16* buf) {
        #pragma unroll
        for (int half = 0; half < 2; ++half) {
            int kq = kk_f + half * 8;
            v8h pack;
            #pragma unroll
            for (int u = 0; u < 8; ++u) {
                int2 e = ldsK[k0 + kq + u];
                float val = 0.f;
                if ((vmask >> e.y) & 1) val = xin[baseF + e.x];
                pack[u] = (_Float16)val;
            }
            *(v8h*)(buf + (size_t)nn_f * 40 + kq) = pack;
        }
    };

    // --- A fragment source (global, row-major f16, stride Kpad) -------------
    const int aRow = mTile + (lane & 15);
    const _Float16* aPtr = Aw + (size_t)aRow * Kpad + ((lane >> 4) << 4);

    // --- B fragment offset within an LDS buffer -----------------------------
    const int bOff = (ni * 64 + (lane & 15)) * 40 + ((lane >> 4) << 4);

    v8f acc0 = {}, acc1 = {}, acc2 = {}, acc3 = {};

    const int nch = Kpad >> 5;
    stage(0, ldsB0);                       // prologue: chunk 0 into buffer 0

    for (int j = 0; j < nch; ++j) {
        __syncthreads();
        const _Float16* cur = (j & 1) ? ldsB1 : ldsB0;
        if (j + 1 < nch)                   // overlap next gathers with WMMAs
            stage((j + 1) << 5, (j & 1) ? ldsB0 : ldsB1);

        const int k0 = j << 5;
        // A fragment: lanes 0-15 hold K k0+0..15 (row lane&15),
        //             lanes 16-31 hold K k0+16..31
        v16h afrag = make16(*(const v8h*)(aPtr + k0),
                            *(const v8h*)(aPtr + k0 + 8));

        // B fragments: lane&15 = column, lane>>4 selects K 0-15 / 16-31
        const _Float16* bp = cur + bOff;
        v16h b0 = make16(*(const v8h*)(bp),           *(const v8h*)(bp + 8));
        v16h b1 = make16(*(const v8h*)(bp + 16 * 40), *(const v8h*)(bp + 16 * 40 + 8));
        v16h b2 = make16(*(const v8h*)(bp + 32 * 40), *(const v8h*)(bp + 32 * 40 + 8));
        v16h b3 = make16(*(const v8h*)(bp + 48 * 40), *(const v8h*)(bp + 48 * 40 + 8));

        acc0 = __builtin_amdgcn_wmma_f32_16x16x32_f16(false, afrag, false, b0,
                                                      (short)0, acc0, false, false);
        acc1 = __builtin_amdgcn_wmma_f32_16x16x32_f16(false, afrag, false, b1,
                                                      (short)0, acc1, false, false);
        acc2 = __builtin_amdgcn_wmma_f32_16x16x32_f16(false, afrag, false, b2,
                                                      (short)0, acc2, false, false);
        acc3 = __builtin_amdgcn_wmma_f32_16x16x32_f16(false, afrag, false, b3,
                                                      (short)0, acc3, false, false);

        if (k0 + 32 < Kpad)
            __builtin_prefetch(aPtr + k0 + 32, 0, 1);   // global_prefetch_b8
    }

    // --- writeback: C layout lane<16 -> M=r, lane>=16 -> M=r+8; N=lane&15 ---
    const int nLoc   = lane & 15;
    const int laneHi = lane >> 4;
    #pragma unroll
    for (int f = 0; f < 4; ++f) {
        const v8f& acc = (f == 0) ? acc0 : (f == 1) ? acc1 : (f == 2) ? acc2 : acc3;
        int n  = nTile0 + ni * 64 + f * 16 + nLoc;
        int ow = n & (OW - 1);
        int oh = (n >> low) & (OW - 1);
        int b  = n >> lohw;
        #pragma unroll
        for (int r = 0; r < 8; ++r) {
            int m = mTile + laneHi * 8 + r;
            int part = (m >= cout) ? 1 : 0;
            int ch = m - part * cout;
            size_t o = ((size_t)(b * cout + ch) << lohw) + ((size_t)oh << low) + ow;
            yout[o * 2 + part] = acc[r];
        }
    }
}

// ---------------------------------------------------------------------------
// Deterministic per-sample |x| statistics: stage 1 (per-WG partials)
// ---------------------------------------------------------------------------
__global__ __launch_bounds__(256) void stats_partial_kernel(
    const float* __restrict__ y, int perSample, int P, float* __restrict__ partials)
{
    __shared__ float s1[256], s2[256];
    int t = threadIdx.x;
    int b = blockIdx.x / P;
    int p = blockIdx.x - b * P;
    const float2* yb = (const float2*)y + (size_t)b * perSample;
    float s = 0.f, q = 0.f;
    for (int i = p * 256 + t; i < perSample; i += P * 256) {
        float2 v = yb[i];
        float m = sqrtf(v.x * v.x + v.y * v.y);
        s += m; q += m * m;
    }
    s1[t] = s; s2[t] = q;
    __syncthreads();
    for (int off = 128; off > 0; off >>= 1) {
        if (t < off) { s1[t] += s1[t + off]; s2[t] += s2[t + off]; }
        __syncthreads();
    }
    if (t == 0) {
        partials[blockIdx.x * 2]     = s1[0];
        partials[blockIdx.x * 2 + 1] = s2[0];
    }
}

// Stage 2: fixed-order sum -> mean, invstd per sample
__global__ void stats_final_kernel(const float* __restrict__ partials, int P,
                                   float invN, float* __restrict__ stats) {
    int t = threadIdx.x;
    if (t < 8) {
        float s = 0.f, q = 0.f;
        for (int j = 0; j < P; ++j) {
            s += partials[(t * P + j) * 2];
            q += partials[(t * P + j) * 2 + 1];
        }
        float mean = s * invN;
        float var  = fmaxf(q * invN - mean * mean, 0.f);
        stats[t * 2]     = mean;
        stats[t * 2 + 1] = rsqrtf(var + 1e-8f);
    }
}

// ---------------------------------------------------------------------------
// Input global-LN (mag norm, keep phase, complex gamma/beta) -> ws buffer
// ---------------------------------------------------------------------------
__global__ __launch_bounds__(256) void norm_apply_kernel(
    const float* __restrict__ xin, const float* __restrict__ gam,
    const float* __restrict__ bet, const float* __restrict__ stats,
    float* __restrict__ out, int lps, int lhw, int cmask)
{
    size_t i = (size_t)blockIdx.x * 256 + threadIdx.x;
    int b = (int)(i >> lps);
    int c = (int)(i >> lhw) & cmask;
    float2 v = ((const float2*)xin)[i];
    float mean = stats[b * 2], inv = stats[b * 2 + 1];
    float mag = sqrtf(v.x * v.x + v.y * v.y);
    float nm = (mag - mean) * inv;
    float2 yn;
    if (mag > 0.f) { float s = nm / mag; yn.x = v.x * s; yn.y = v.y * s; }
    else           { yn.x = nm; yn.y = 0.f; }           // angle(0)=0 -> phase=1
    float2 g  = ((const float2*)gam)[c];
    float2 bb = ((const float2*)bet)[c];
    ((float2*)out)[i] = cmulacc(yn, g, bb);
}

// ---------------------------------------------------------------------------
// FiLM: per-sample h = z@fc1^T+b1 ; p = h@fc2^T+b2 (complex). One block per b.
// p stored at wsP[b*512 + i] (float2).
// ---------------------------------------------------------------------------
__global__ __launch_bounds__(256) void film_kernel(
    const float* __restrict__ z,  const float* __restrict__ f1w,
    const float* __restrict__ f1b, const float* __restrict__ f2w,
    const float* __restrict__ f2b, float* __restrict__ wsP,
    int hdim, int twoCout)
{
    __shared__ float2 zs[128];
    __shared__ float2 hs[512];
    int b = blockIdx.x, t = threadIdx.x;
    if (t < 128) zs[t] = ((const float2*)z)[b * 128 + t];
    __syncthreads();
    for (int j = t; j < hdim; j += 256) {
        float2 acc = ((const float2*)f1b)[j];
        const float2* wr = (const float2*)f1w + (size_t)j * 128;
        for (int d = 0; d < 128; ++d) acc = cmulacc(zs[d], wr[d], acc);
        hs[j] = acc;
    }
    __syncthreads();
    for (int i = t; i < twoCout; i += 256) {
        float2 acc = ((const float2*)f2b)[i];
        const float2* wr = (const float2*)f2w + (size_t)i * hdim;
        for (int j = 0; j < hdim; ++j) acc = cmulacc(hs[j], wr[j], acc);
        ((float2*)wsP)[b * 512 + i] = acc;
    }
}

// ---------------------------------------------------------------------------
// Fused globLN + FiLM + CReLU, in-place on an output segment.
// ---------------------------------------------------------------------------
__global__ __launch_bounds__(256) void post_kernel(
    float* __restrict__ y, const float* __restrict__ lng,
    const float* __restrict__ lnb, const float* __restrict__ wsP,
    const float* __restrict__ stats, int lps, int lohw, int cout)
{
    size_t i = (size_t)blockIdx.x * 256 + threadIdx.x;
    int b = (int)(i >> lps);
    int c = (int)(i >> lohw) & (cout - 1);
    float2* y2 = (float2*)y;
    float2 v = y2[i];
    float mean = stats[b * 2], inv = stats[b * 2 + 1];
    float mag = sqrtf(v.x * v.x + v.y * v.y);
    float nm = (mag - mean) * inv;
    float2 yn;
    if (mag > 0.f) { float s = nm / mag; yn.x = v.x * s; yn.y = v.y * s; }
    else           { yn.x = nm; yn.y = 0.f; }
    float2 g  = ((const float2*)lng)[c];
    float2 bb = ((const float2*)lnb)[c];
    float2 tv = cmulacc(yn, g, bb);
    float2 fg = ((const float2*)wsP)[b * 512 + c];        fg.x += 1.f;
    float2 fb = ((const float2*)wsP)[b * 512 + cout + c];
    float2 o = cmulacc(tv, fg, fb);
    y2[i] = make_float2(fmaxf(o.x, 0.f), fmaxf(o.y, 0.f));
}

// ---------------------------------------------------------------------------
// Host side
// ---------------------------------------------------------------------------
extern "C" void kernel_launch(void* const* d_in, const int* in_sizes, int n_in,
                              void* d_out, int out_size, void* d_ws, size_t ws_size,
                              hipStream_t stream) {
    (void)in_sizes; (void)n_in; (void)out_size; (void)ws_size;

    const float* x  = (const float*)d_in[0];   // complex (8,2,256,256)
    const float* z  = (const float*)d_in[1];   // complex (8,128)
    const float* gg = (const float*)d_in[2];   // globln_gamma (2)
    const float* gb = (const float*)d_in[3];   // globln_beta  (2)

    // Block configs
    const int cinA[5]    = {2, 32, 64, 128, 256};
    const int lcA[5]     = {1, 5, 6, 7, 8};
    const int coutA[5]   = {32, 64, 128, 256, 256};
    const int lcoutA[5]  = {5, 6, 7, 8, 8};
    const int strideA[5] = {1, 2, 2, 2, 2};
    const int liwA[5]    = {8, 8, 7, 6, 5};   // log2 input H (=W)
    const int lowA[5]    = {8, 7, 6, 5, 4};   // log2 output H (=W)
    const int hdimA[5]   = {128, 128, 256, 512, 512};

    // Workspace carve-up
    char* ws = (char*)d_ws;
    float*     xnorm    = (float*)ws;                                // 8 MB
    _Float16*  Aw       = (_Float16*)(ws + (8u << 20));              // <= 4.5 MB
    float*     partials = (float*)(ws + (13u << 20));                // 8 KB
    float*     stats    = (float*)(ws + (13u << 20) + (64u << 10));
    float*     wsP      = (float*)(ws + (13u << 20) + (128u << 10)); // 32 KB
    int2*      ktab     = (int2*)(ws + (13u << 20) + (256u << 10));  // <= 36 KB

    const int P = 128;
    float* outF = (float*)d_out;

    // ---- input global-LN -> xnorm -----------------------------------------
    {
        int perSample = 2 << 16;                 // IC*H*W = 131072
        stats_partial_kernel<<<8 * P, 256, 0, stream>>>(x, perSample, P, partials);
        stats_final_kernel<<<1, 64, 0, stream>>>(partials, P, 1.f / perSample, stats);
        int total = 8 * perSample;               // 1,048,576
        norm_apply_kernel<<<total / 256, 256, 0, stream>>>(
            x, gg, gb, stats, xnorm, /*lps=*/17, /*lhw=*/16, /*cmask=*/1);
    }

    // ---- output segment offsets (complex elements) ------------------------
    size_t segOff[6]; segOff[0] = 0;
    for (int i = 0; i < 5; ++i) {
        size_t s = (size_t)8 * coutA[i] << (2 * lowA[i]);
        segOff[i + 1] = segOff[i] + s;
    }

    const float* xprev = xnorm;
    for (int i = 0; i < 5; ++i) {
        const float* cw  = (const float*)d_in[4 + 7 * i + 0];
        const float* lng = (const float*)d_in[4 + 7 * i + 1];
        const float* lnb = (const float*)d_in[4 + 7 * i + 2];
        const float* f1w = (const float*)d_in[4 + 7 * i + 3];
        const float* f1b = (const float*)d_in[4 + 7 * i + 4];
        const float* f2w = (const float*)d_in[4 + 7 * i + 5];
        const float* f2b = (const float*)d_in[4 + 7 * i + 6];

        const int lc = lcA[i], cout = coutA[i];
        const int liw = liwA[i], low = lowA[i];
        const int lihw = 2 * liw, lohw = 2 * low;
        const int K = 18 * cinA[i];
        const int Kpad = (K + 31) & ~31;
        const int M = 2 * cout;
        const int N = 8 << lohw;

        float* yseg = outF + segOff[i] * 2;

        // weights -> f16 GEMM A matrix; k-decode table
        wtrans_kernel<<<dim3((unsigned)(Kpad / 64), (unsigned)M), 64, 0, stream>>>(
            cw, Aw, lc, cout, K, Kpad);
        ktab_kernel<<<(unsigned)(Kpad / 64), 64, 0, stream>>>(
            ktab, lc, liw, lihw, K, Kpad);

        // FiLM coefficients for this block
        film_kernel<<<8, 256, 0, stream>>>(z, f1w, f1b, f2w, f2b, wsP,
                                           hdimA[i], 2 * cout);

        // implicit-GEMM complex conv (WMMA) -> raw complex into d_out segment
        conv_wmma_kernel<<<dim3((unsigned)(N / 128), (unsigned)(M / 64)), 256, 0, stream>>>(
            xprev, Aw, ktab, yseg, lc, strideA[i], liw, lihw, low, lohw, cout, Kpad);

        // per-sample magnitude stats of the conv output
        int perSample = cout << lohw;
        stats_partial_kernel<<<8 * P, 256, 0, stream>>>(yseg, perSample, P, partials);
        stats_final_kernel<<<1, 64, 0, stream>>>(partials, P, 1.f / perSample, stats);

        // fused globLN + FiLM + CReLU, in place
        int lps = lcoutA[i] + lohw;
        size_t total = (size_t)8 * perSample;
        post_kernel<<<(unsigned)(total / 256), 256, 0, stream>>>(
            yseg, lng, lnb, wsP, stats, lps, lohw, cout);

        xprev = yseg;   // next block consumes this activation
    }
}
